// MSDeformAttn_64287070487202
// MI455X (gfx1250) — matrix operand order repaired and checked by reading, not compiled, not run on hardware
//
#include <hip/hip_runtime.h>

// ---------------------------------------------------------------------------
// MSDeformAttn forward for gfx1250 (MI455X), wave32 + WMMA + async-LDS DMA.
//   0) prep: f32->f16 activations; weights transposed+converted to [K][N] f16
//   1) value = input_flatten @ W_value^T + b    (WMMA f16 GEMM, f32 accum)
//   2) off   = query @ W_off^T  + b_off         (WMMA GEMM)
//   3) attn  = query @ W_attn^T + b_attn        (WMMA GEMM)
//   4) fused softmax + bilinear sampling        (1 wave per (n,q,head)) -> f16
//   5) out   = sampled @ W_out^T + b_out        (WMMA GEMM)
// GEMM inner loop stages BOTH tiles with global_load_async_to_lds_b128:
// no VALU conversions, no staging stores -- async DMA + ds_load_b128 + wmma.
// ---------------------------------------------------------------------------

typedef __attribute__((ext_vector_type(16))) _Float16 v16h;
typedef __attribute__((ext_vector_type(8)))  _Float16 v8h;
typedef __attribute__((ext_vector_type(8)))  float    v8f;
typedef __attribute__((ext_vector_type(4)))  float    v4f;
typedef __attribute__((ext_vector_type(4)))  int      v4i;

#define D_MODEL   256
#define N_HEADS   8
#define LEN_IN    19947
#define LQ        8000
#define NBATCH    4

#define TILE_M 128
#define TILE_N 128
#define TILE_K 32
#define LDA_P  40     // padded LDS row stride (f16 elems) for A tile (32 cols)
#define LDB_P  136    // padded LDS row stride (f16 elems) for B tile (128 cols)

#if defined(__AMDGCN__) && __has_builtin(__builtin_amdgcn_global_load_async_to_lds_b128)
#define HAS_ASYNC_LDS 1
typedef const __attribute__((address_space(1))) void* as1_cvptr;
typedef __attribute__((address_space(3))) void*       as3_vptr;
typedef __attribute__((address_space(1))) v4i*        as1_v4i_ptr;
typedef __attribute__((address_space(3))) v4i*        as3_v4i_ptr;
#else
#define HAS_ASYNC_LDS 0
#endif

static __device__ __forceinline__ void wait_async_all()
{
#if defined(__AMDGCN__)
#if __has_builtin(__builtin_amdgcn_s_wait_asynccnt)
    __builtin_amdgcn_s_wait_asynccnt(0);
#else
    asm volatile("s_wait_asynccnt 0x0" ::: "memory");
#endif
#endif
}

static __device__ __forceinline__ void async_copy_b128(const _Float16* src, _Float16* lds_dst)
{
#if HAS_ASYNC_LDS
    as1_v4i_ptr g = (as1_v4i_ptr)(as1_cvptr)(const void*)src;  // addrspacecast + retype
    as3_v4i_ptr l = (as3_v4i_ptr)(as3_vptr)(void*)lds_dst;
    __builtin_amdgcn_global_load_async_to_lds_b128(g, l, 0, 0);
#else
    *(v4i*)lds_dst = *(const v4i*)src;
#endif
}

union Frag16 { v16h v; v8h h[2]; };

// C[M,N] = A16[M,K] * Bt16[K,N] + bias[N]     (f16 in, f32 accum/out)
__global__ __launch_bounds__(256)
void msda_gemm_f16_kernel(const _Float16* __restrict__ A16,
                          const _Float16* __restrict__ Bt16,
                          const float* __restrict__ bias,
                          float* __restrict__ C,
                          int M, int N, int K)
{
    __shared__ __attribute__((aligned(16))) _Float16 sA[TILE_M * LDA_P];
    __shared__ __attribute__((aligned(16))) _Float16 sB[TILE_K * LDB_P];

    const int tid  = threadIdx.x;
    const int lane = tid & 31;
    const int wv   = tid >> 5;     // 8 waves
    const int mw   = wv >> 2;      // 0..1 -> 64-row strip
    const int nw   = wv & 3;       // 0..3 -> 32-col strip
    const int bm   = blockIdx.x * TILE_M;
    const int bn   = blockIdx.y * TILE_N;

    v8f acc[4][2];
#pragma unroll
    for (int i = 0; i < 4; ++i)
#pragma unroll
        for (int j = 0; j < 2; ++j)
            acc[i][j] = (v8f){0.f, 0.f, 0.f, 0.f, 0.f, 0.f, 0.f, 0.f};

    const int m16 = lane & 15;
    const int kb  = (lane >> 4) << 3;   // 0 or 8: K-group base for A fragment

    // A staging map: thread -> (row, 32B half-row). OOB rows clamped; their
    // garbage only reaches C rows >= M which the epilogue never stores.
    const int ar_  = tid >> 1;
    const int ahf  = (tid & 1) << 4;          // 0 or 16 f16 elems (32B)
    // B staging map: thread -> (k row, 32B segment of 256B row)
    const int bk_  = tid >> 3;                // 0..31
    const int bseg = (tid & 7) << 4;          // 0,16,...,112 f16 elems

    for (int k0 = 0; k0 < K; k0 += TILE_K) {
        // ---- stage A tile (f16) via async DMA: 128 rows x 64B ----
        {
            const int gr = min(bm + ar_, M - 1);
            const _Float16* src = A16 + (size_t)gr * K + k0 + ahf;
            _Float16*       dst = &sA[ar_ * LDA_P + ahf];
            async_copy_b128(src + 0, dst + 0);
            async_copy_b128(src + 8, dst + 8);
            if (k0 + TILE_K < K)
                __builtin_prefetch(src + TILE_K, 0, 3);   // global_prefetch_b8
        }
        // ---- stage B tile (f16, pre-transposed [K][N]) via async DMA ----
        {
            const _Float16* src = Bt16 + (size_t)(k0 + bk_) * N + bn + bseg;
            _Float16*       dst = &sB[bk_ * LDB_P + bseg];
            async_copy_b128(src + 0, dst + 0);
            async_copy_b128(src + 8, dst + 8);
            if (k0 + TILE_K < K)
                __builtin_prefetch(src + (size_t)TILE_K * N, 0, 3);
        }
        wait_async_all();
        __syncthreads();

        // ---- fragments per documented gfx1250 WMMA layouts ----
        // A 16x32 f16: lane m=lane&15; lanes 0-15 hold K 0-7 & 16-23,
        //              lanes 16-31 hold K 8-15 & 24-31 (contiguous halves).
        Frag16 afr[4];
#pragma unroll
        for (int mt = 0; mt < 4; ++mt) {
            const _Float16* ar = &sA[(mw * 64 + mt * 16 + m16) * LDA_P];
            afr[mt].h[0] = *(const v8h*)(ar + kb);
            afr[mt].h[1] = *(const v8h*)(ar + kb + 16);
        }
        // B 32x16 f16: lane n=lane&15; same K order as A (kb..kb+7, kb+16..kb+23).
        Frag16 bfr[2];
#pragma unroll
        for (int nt = 0; nt < 2; ++nt) {
            const _Float16* br = &sB[nw * 32 + nt * 16 + m16];
#pragma unroll
            for (int s = 0; s < 8; ++s) {
                bfr[nt].v[s]     = br[(kb + s) * LDB_P];
                bfr[nt].v[s + 8] = br[(kb + 16 + s) * LDB_P];
            }
        }
#pragma unroll
        for (int mt = 0; mt < 4; ++mt)
#pragma unroll
            for (int nt = 0; nt < 2; ++nt)
                acc[mt][nt] = __builtin_amdgcn_wmma_f32_16x16x32_f16(
                    false, afr[mt].v, false, bfr[nt].v,
                    (short)0, acc[mt][nt], false, false);
        __syncthreads();
    }

    // ---- epilogue: C/D layout lane n=lane&15, VGPR r -> M = r + 8*(lane>=16)
    const int ncol = lane & 15;
    const int mhi  = (lane >> 4) << 3;
#pragma unroll
    for (int mt = 0; mt < 4; ++mt) {
#pragma unroll
        for (int nt = 0; nt < 2; ++nt) {
            const int gc = bn + nw * 32 + nt * 16 + ncol;
            const float bv = bias[gc];
#pragma unroll
            for (int r = 0; r < 8; ++r) {
                const int gr = bm + mw * 64 + mt * 16 + mhi + r;
                if (gr < M)
                    C[(size_t)gr * N + gc] = acc[mt][nt][r] + bv;
            }
        }
    }
}

// ---- prep: f32 -> f16 (count must be a multiple of 8) ----------------------
__global__ __launch_bounds__(256)
void cvt_f16_kernel(const float* __restrict__ src, _Float16* __restrict__ dst, int count)
{
    const int idx = (blockIdx.x * 256 + threadIdx.x) * 8;
    if (idx < count) {
        v4f f0 = *(const v4f*)(src + idx);
        v4f f1 = *(const v4f*)(src + idx + 4);
        v8h h;
#pragma unroll
        for (int j = 0; j < 4; ++j) {
            h[j]     = (_Float16)f0[j];
            h[j + 4] = (_Float16)f1[j];
        }
        *(v8h*)(dst + idx) = h;
    }
}

// ---- prep: W[N][K] f32 -> Wt[K][N] f16 (tiny matrices) ---------------------
__global__ __launch_bounds__(256)
void transpose_cvt_kernel(const float* __restrict__ W, _Float16* __restrict__ Wt,
                          int N, int K)
{
    const int idx = blockIdx.x * 256 + threadIdx.x;
    if (idx < N * K) {
        const int n = idx / K;
        const int k = idx - n * K;
        Wt[(size_t)k * N + n] = (_Float16)W[idx];
    }
}

// ---------------------------------------------------------------------------
// Fused softmax + bilinear sampling. One wave per (n, q, head); lane = channel.
// Output written directly as f16 (feeds the final WMMA GEMM).
// ---------------------------------------------------------------------------
__global__ __launch_bounds__(256)
void msda_sample_kernel(const float* __restrict__ value,
                        const float* __restrict__ offb,
                        const float* __restrict__ attnb,
                        const float* __restrict__ refp,
                        _Float16* __restrict__ outb)
{
    __shared__ __attribute__((aligned(16))) float sm[8][16][8]; // [wave][point][w00..w11, i00..i11]

    const int tid  = threadIdx.x;
    const int lane = tid & 31;
    const int wv   = tid >> 5;
    const int gw   = blockIdx.x * 8 + wv;      // 0 .. N*LQ*N_HEADS-1
    const int h    = gw & 7;
    const int nq   = gw >> 3;                  // 0 .. N*LQ-1
    const int n    = nq / LQ;

    // ---- softmax over 16 (L*P) logits: point i lives on lane i (mirrored) ----
    const int i = lane & 15;
    const int l = i >> 2;                      // level
    float a = attnb[(size_t)nq * 128 + h * 16 + i];
    float mx = a;
#pragma unroll
    for (int m = 1; m <= 8; m <<= 1) mx = fmaxf(mx, __shfl_xor(mx, m, 32));
    float e = __expf(a - mx);
    float s = e;
#pragma unroll
    for (int m = 1; m <= 8; m <<= 1) s += __shfl_xor(s, m, 32);
    const float aw = e / s;

    // ---- geometry for point i ----
    const float ox = offb[(size_t)nq * 256 + h * 32 + i * 2 + 0];
    const float oy = offb[(size_t)nq * 256 + h * 32 + i * 2 + 1];
    const float rx = refp[((size_t)nq * 4 + l) * 2 + 0];
    const float ry = refp[((size_t)nq * 4 + l) * 2 + 1];

    const int Wl = (l == 0) ? 150 : (l == 1) ? 75 : (l == 2) ? 38 : 19;
    const int Hl = (l == 0) ? 100 : (l == 1) ? 50 : (l == 2) ? 25 : 13;
    const int Sl = (l == 0) ? 0 : (l == 1) ? 15000 : (l == 2) ? 18750 : 19700;
    const float fw = (float)Wl, fh = (float)Hl;

    const float x  = (rx + ox / fw) * fw - 0.5f;
    const float y  = (ry + oy / fh) * fh - 0.5f;
    const float xf = floorf(x), yf = floorf(y);
    const float lx = x - xf,  ly = y - yf;
    const int x0 = (int)xf, y0 = (int)yf;
    const int x1 = x0 + 1,  y1 = y0 + 1;

    const float vx0 = (x0 >= 0 && x0 < Wl) ? 1.f : 0.f;
    const float vx1 = (x1 >= 0 && x1 < Wl) ? 1.f : 0.f;
    const float vy0 = (y0 >= 0 && y0 < Hl) ? 1.f : 0.f;
    const float vy1 = (y1 >= 0 && y1 < Hl) ? 1.f : 0.f;
    const int cx0 = min(max(x0, 0), Wl - 1);
    const int cx1 = min(max(x1, 0), Wl - 1);
    const int cy0 = min(max(y0, 0), Hl - 1);
    const int cy1 = min(max(y1, 0), Hl - 1);

    const int i00 = Sl + cy0 * Wl + cx0;
    const int i01 = Sl + cy0 * Wl + cx1;
    const int i10 = Sl + cy1 * Wl + cx0;
    const int i11 = Sl + cy1 * Wl + cx1;
    const float w00 = aw * (1.f - lx) * (1.f - ly) * vx0 * vy0;
    const float w01 = aw * lx * (1.f - ly) * vx1 * vy0;
    const float w10 = aw * (1.f - lx) * ly * vx0 * vy1;
    const float w11 = aw * lx * ly * vx1 * vy1;

    if (lane < 16) {
        sm[wv][i][0] = w00;
        sm[wv][i][1] = w01;
        sm[wv][i][2] = w10;
        sm[wv][i][3] = w11;
        sm[wv][i][4] = __int_as_float(i00);
        sm[wv][i][5] = __int_as_float(i01);
        sm[wv][i][6] = __int_as_float(i10);
        sm[wv][i][7] = __int_as_float(i11);
    }
    __syncthreads();

    // ---- gather + weighted accumulate, lane = channel d (coalesced 128B) ----
    const float* vb = value + (size_t)n * LEN_IN * 256 + h * 32 + lane;
    float acc = 0.f;
#pragma unroll
    for (int p = 0; p < 16; ++p) {
        const float W00 = sm[wv][p][0];
        const float W01 = sm[wv][p][1];
        const float W10 = sm[wv][p][2];
        const float W11 = sm[wv][p][3];
        const int j00 = __float_as_int(sm[wv][p][4]);
        const int j01 = __float_as_int(sm[wv][p][5]);
        const int j10 = __float_as_int(sm[wv][p][6]);
        const int j11 = __float_as_int(sm[wv][p][7]);
        acc += W00 * vb[(size_t)j00 * 256];
        acc += W01 * vb[(size_t)j01 * 256];
        acc += W10 * vb[(size_t)j10 * 256];
        acc += W11 * vb[(size_t)j11 * 256];
    }
    outb[(size_t)nq * 256 + h * 32 + lane] = (_Float16)acc;
}

// ---------------------------------------------------------------------------
extern "C" void kernel_launch(void* const* d_in, const int* in_sizes, int n_in,
                              void* d_out, int out_size, void* d_ws, size_t ws_size,
                              hipStream_t stream)
{
    const float* query   = (const float*)d_in[0];   // (4,8000,256)
    const float* refpts  = (const float*)d_in[1];   // (4,8000,4,2)
    const float* in_flat = (const float*)d_in[2];   // (4,19947,256)
    // d_in[3]/d_in[4]: spatial shapes / level starts (static, baked into kernel)
    const float* W_value = (const float*)d_in[5];
    const float* b_value = (const float*)d_in[6];
    const float* W_off   = (const float*)d_in[7];
    const float* b_off   = (const float*)d_in[8];
    const float* W_attn  = (const float*)d_in[9];
    const float* b_attn  = (const float*)d_in[10];
    const float* W_out   = (const float*)d_in[11];
    const float* b_out   = (const float*)d_in[12];
    float* out = (float*)d_out;

    const int M_VAL = NBATCH * LEN_IN;   // 79788
    const int M_Q   = NBATCH * LQ;       // 32000

    // ---- workspace carve-up (all offsets 16B-aligned) ----
    char* ws = (char*)d_ws;
    size_t o = 0;
    float*     value_ws = (float*)(ws + o);     o += (size_t)M_VAL * 256 * 4;  // 81.7MB
    float*     off_ws   = (float*)(ws + o);     o += (size_t)M_Q * 256 * 4;    // 32.8MB
    float*     attn_ws  = (float*)(ws + o);     o += (size_t)M_Q * 128 * 4;    // 16.4MB
    _Float16*  inflat16 = (_Float16*)(ws + o);  o += (size_t)M_VAL * 256 * 2;  // 40.9MB
    _Float16*  query16  = (_Float16*)(ws + o);  o += (size_t)M_Q * 256 * 2;    // 16.4MB
    _Float16*  tmp16    = (_Float16*)(ws + o);  o += (size_t)M_Q * 256 * 2;    // 16.4MB
    _Float16*  wt_val   = (_Float16*)(ws + o);  o += (size_t)256 * 256 * 2;
    _Float16*  wt_off   = (_Float16*)(ws + o);  o += (size_t)256 * 256 * 2;
    _Float16*  wt_attn  = (_Float16*)(ws + o);  o += (size_t)128 * 256 * 2;
    _Float16*  wt_out   = (_Float16*)(ws + o);  o += (size_t)256 * 256 * 2;

    const dim3 blk(256);

    // ---- prep: activations f32->f16; weights transpose+convert ----
    {
        const int c1 = M_VAL * 256;
        cvt_f16_kernel<<<dim3((c1 / 8 + 255) / 256), blk, 0, stream>>>(in_flat, inflat16, c1);
        const int c2 = M_Q * 256;
        cvt_f16_kernel<<<dim3((c2 / 8 + 255) / 256), blk, 0, stream>>>(query, query16, c2);
        transpose_cvt_kernel<<<dim3((256 * 256 + 255) / 256), blk, 0, stream>>>(W_value, wt_val, 256, 256);
        transpose_cvt_kernel<<<dim3((256 * 256 + 255) / 256), blk, 0, stream>>>(W_off, wt_off, 256, 256);
        transpose_cvt_kernel<<<dim3((128 * 256 + 255) / 256), blk, 0, stream>>>(W_attn, wt_attn, 128, 256);
        transpose_cvt_kernel<<<dim3((256 * 256 + 255) / 256), blk, 0, stream>>>(W_out, wt_out, 256, 256);
    }

    // 1) value projection
    msda_gemm_f16_kernel<<<dim3((M_VAL + TILE_M - 1) / TILE_M, D_MODEL / TILE_N),
                           blk, 0, stream>>>(inflat16, wt_val, b_value, value_ws,
                                             M_VAL, D_MODEL, D_MODEL);
    // 2) sampling-offset projection (256 outputs)
    msda_gemm_f16_kernel<<<dim3(M_Q / TILE_M, 256 / TILE_N),
                           blk, 0, stream>>>(query16, wt_off, b_off, off_ws,
                                             M_Q, 256, D_MODEL);
    // 3) attention-logit projection (128 outputs)
    msda_gemm_f16_kernel<<<dim3(M_Q / TILE_M, 128 / TILE_N),
                           blk, 0, stream>>>(query16, wt_attn, b_attn, attn_ws,
                                             M_Q, 128, D_MODEL);
    // 4) fused softmax + multiscale bilinear sampling -> f16
    msda_sample_kernel<<<dim3((NBATCH * LQ * N_HEADS) / 8),
                         blk, 0, stream>>>(value_ws, off_ws, attn_ws, refpts, tmp16);
    // 5) output projection
    msda_gemm_f16_kernel<<<dim3(M_Q / TILE_M, D_MODEL / TILE_N),
                           blk, 0, stream>>>(tmp16, wt_out, b_out, out,
                                             M_Q, D_MODEL, D_MODEL);
}
